// Mini_pointgnn_v13_67310727463247
// MI455X (gfx1250) — matrix operand order repaired and checked by reading, not compiled, not run on hardware
//
#include <hip/hip_runtime.h>

#define DEVINL __device__ __forceinline__

typedef __attribute__((ext_vector_type(16))) __bf16 bf16x16;
typedef __attribute__((ext_vector_type(8)))  float  f32x8;

constexpr int N    = 524288;
constexpr int N1   = 65536;
constexpr int N2   = 16384;
constexpr int E2   = 524288;
constexpr int NCLS = 20;

constexpr int WPB   = 4;         // waves per block
constexpr int BLOCK = 32 * WPB;
constexpr int STG   = 100;       // LDS stage row stride (floats), 16 x 96 payload (even => 8B pairs)

DEVINL __bf16 tobf(float x) { return (__bf16)x; }

// ---- A fragment: 16x32 bf16 tile from LDS stage (row-major, stride STG) ----
// ISA layout: lanes0-15 row=lane, V0..3 K=2v..2v+1, V4..7 K=16+2(v-4)..;
//             lanes16-31 same rows, K offset by +8 (V0..3) / +8 (V4..7).
DEVINL bf16x16 load_A_frag(const float* xs, int kbase, int lane) {
  int row = lane & 15, half = (lane >> 4) & 1;
  bf16x16 a;
#pragma unroll
  for (int v = 0; v < 8; ++v) {
    int k0 = (v < 4) ? ((half ? 8 : 0) + 2 * v) : ((half ? 24 : 16) + 2 * (v - 4));
    float2 p = *(const float2*)(xs + row * STG + kbase + k0);  // 8B-aligned ds_load_b64
    a[2 * v]     = tobf(p.x);
    a[2 * v + 1] = tobf(p.y);
  }
  return a;
}

// ---- B fragment: pre-packed bf16, one 32B-aligned 8-dword record per lane ----
DEVINL bf16x16 load_B_pre(const unsigned* __restrict__ frag, int kt, int nt, int NT, int lane) {
  const unsigned* p = frag + ((size_t)(kt * NT + nt) * 32 + lane) * 8;
  union { uint4 q[2]; bf16x16 v; } U;
  U.q[0] = *(const uint4*)p;        // global_load_b128
  U.q[1] = *(const uint4*)(p + 4);  // global_load_b128
  return U.v;
}

// ---- one dense layer on a 16-row tile: acc[NT] covers NT*16 output cols ----
template <int NT>
DEVINL void mlp_layer(const float* xs, const unsigned* __restrict__ frag,
                      const float* __restrict__ bias,
                      int ktiles, bool relu, int lane, f32x8* acc) {
  int col = lane & 15;
#pragma unroll
  for (int j = 0; j < NT; ++j) {
    float bv = bias[j * 16 + col];
    f32x8 c;
#pragma unroll
    for (int v = 0; v < 8; ++v) c[v] = bv;
    for (int kt = 0; kt < ktiles; ++kt) {
      bf16x16 a = load_A_frag(xs, kt * 32, lane);
      bf16x16 b = load_B_pre(frag, kt, j, NT, lane);
      c = __builtin_amdgcn_wmma_f32_16x16x32_bf16(false, a, false, b, (short)0, c, false, false);
    }
    if (relu) {
#pragma unroll
      for (int v = 0; v < 8; ++v) c[v] = fmaxf(c[v], 0.f);
    }
    acc[j] = c;
  }
}

// ---- write C/D fragments back to the LDS stage (chained layers) ----
// C layout: VGPR v -> row = v + 8*(lane>=16), col = lane&15
template <int NT>
DEVINL void frag_to_lds(float* xs, const f32x8* acc, int lane) {
  int col = lane & 15, half = (lane >> 4) & 1;
#pragma unroll
  for (int j = 0; j < NT; ++j)
#pragma unroll
    for (int v = 0; v < 8; ++v)
      xs[(half * 8 + v) * STG + j * 16 + col] = acc[j][v];
}

template <int KC>
DEVINL void zero_stage(float* xs, int lane) {
#pragma unroll
  for (int i = lane; i < 16 * KC; i += 32) xs[(i / KC) * STG + (i % KC)] = 0.f;
}

DEVINL void atomic_fmax_nonneg(float* addr, float v) {
  atomicMax((unsigned int*)addr, __float_as_uint(v));  // valid: all values >= 0
}

// =====================================================================
__global__ __launch_bounds__(256) void k_fill(float* p, int n, float v) {
  int i = blockIdx.x * blockDim.x + threadIdx.x;
  if (i < n) p[i] = v;
}

// pack W[din][dout] (fp32 row-major) into bf16 B-fragment records
__global__ __launch_bounds__(256) void k_packB(const float* __restrict__ W, int din, int dout,
                                               int KT, int NT, unsigned* __restrict__ frag) {
  int t = blockIdx.x * blockDim.x + threadIdx.x;
  if (t >= KT * NT * 256) return;
  int v = t & 7;
  int lane = (t >> 3) & 31;
  int ft = t >> 8;                 // fragment = kt*NT + nt
  int nt = ft % NT, kt = ft / NT;
  int col = lane & 15, half = (lane >> 4) & 1;
  int k = kt * 32 + ((v < 4) ? ((half ? 8 : 0) + 2 * v) : ((half ? 24 : 16) + 2 * (v - 4)));
  int n = nt * 16 + col;
  float x0 = (k < din && n < dout) ? W[(size_t)k * dout + n] : 0.f;
  float x1 = (k + 1 < din && n < dout) ? W[(size_t)(k + 1) * dout + n] : 0.f;
  union { __bf16 b[2]; unsigned u; } pk;
  pk.b[0] = tobf(x0);
  pk.b[1] = tobf(x1);
  frag[t] = pk.u;
}

__global__ void k_packbias(const float* __restrict__ b, int dout, float* __restrict__ o) {
  int i = threadIdx.x;
  o[i] = (i < dout) ? b[i] : 0.f;
}

// l1: [rem, points - c1[l1_labels]] (din=4) -> 64 relu -> 64 relu, seg-sum into f1[N1,64]
__global__ __launch_bounds__(BLOCK) void k_l1(const float* __restrict__ rem,
                                              const float* __restrict__ pts,
                                              const float* __restrict__ c1,
                                              const int* __restrict__ lab1,
                                              const unsigned* w1, const float* b1,
                                              const unsigned* w2, const float* b2,
                                              float* __restrict__ f1) {
  __shared__ float stage[WPB][16 * STG];
  __shared__ int lab[WPB][16];
  int lane = threadIdx.x & 31, w = threadIdx.x >> 5;
  int r0 = (blockIdx.x * WPB + w) * 16;
  float* xs = stage[w];
  zero_stage<32>(xs, lane);
  if (lane < 16) {
    int r = r0 + lane;
    int lb = lab1[r];
    lab[w][lane] = lb;
    xs[lane * STG + 0] = rem[r];
#pragma unroll
    for (int i = 0; i < 3; ++i) xs[lane * STG + 1 + i] = pts[r * 3 + i] - c1[lb * 3 + i];
  }
  f32x8 acc[4];
  mlp_layer<4>(xs, w1, b1, 1, true, lane, acc);
  frag_to_lds<4>(xs, acc, lane);
  mlp_layer<4>(xs, w2, b2, 2, true, lane, acc);
  int col = lane & 15, half = (lane >> 4) & 1;
#pragma unroll
  for (int j = 0; j < 4; ++j)
#pragma unroll
    for (int v = 0; v < 8; ++v) {
      int row = half * 8 + v;
      atomicAdd(&f1[(size_t)lab[w][row] * 64 + j * 16 + col], acc[j][v]);
    }
}

// l3: [f1, c1 - c2[l2_labels]] (din=67) -> 64 relu -> 64 relu, seg-max into f2[N2,64]
__global__ __launch_bounds__(BLOCK) void k_l3(const float* __restrict__ f1,
                                              const float* __restrict__ c1,
                                              const float* __restrict__ c2,
                                              const int* __restrict__ lab2,
                                              const unsigned* w1, const float* b1,
                                              const unsigned* w2, const float* b2,
                                              float* __restrict__ agg) {
  __shared__ float stage[WPB][16 * STG];
  __shared__ int lab[WPB][16];
  int lane = threadIdx.x & 31, w = threadIdx.x >> 5;
  int r0 = (blockIdx.x * WPB + w) * 16;
  float* xs = stage[w];
  zero_stage<96>(xs, lane);
  if (lane < 16) {
    int r = r0 + lane;
    int lb = lab2[r];
    lab[w][lane] = lb;
#pragma unroll
    for (int i = 0; i < 3; ++i) xs[lane * STG + 64 + i] = c1[r * 3 + i] - c2[lb * 3 + i];
  }
  for (int i = lane; i < 16 * 64; i += 32) {
    int row = i >> 6, k = i & 63;
    xs[row * STG + k] = f1[(size_t)(r0 + row) * 64 + k];
  }
  f32x8 acc[4];
  mlp_layer<4>(xs, w1, b1, 3, true, lane, acc);
  frag_to_lds<4>(xs, acc, lane);
  mlp_layer<4>(xs, w2, b2, 2, true, lane, acc);
  int col = lane & 15, half = (lane >> 4) & 1;
#pragma unroll
  for (int j = 0; j < 4; ++j)
#pragma unroll
    for (int v = 0; v < 8; ++v) {
      int row = half * 8 + v;
      atomic_fmax_nonneg(&agg[(size_t)lab[w][row] * 64 + j * 16 + col], acc[j][v]);
    }
}

// gnn "h": f (64) -> 64 relu -> 3 (no relu) => delta[N2,3]
__global__ __launch_bounds__(BLOCK) void k_delta(const float* __restrict__ f,
                                                 const unsigned* w1, const float* b1,
                                                 const unsigned* w2, const float* b2,
                                                 float* __restrict__ delta) {
  __shared__ float stage[WPB][16 * STG];
  int lane = threadIdx.x & 31, w = threadIdx.x >> 5;
  int r0 = (blockIdx.x * WPB + w) * 16;
  float* xs = stage[w];
  for (int i = lane; i < 16 * 64; i += 32) {
    int row = i >> 6, k = i & 63;
    xs[row * STG + k] = f[(size_t)(r0 + row) * 64 + k];
  }
  f32x8 acc[4];
  mlp_layer<4>(xs, w1, b1, 2, true, lane, acc);
  frag_to_lds<4>(xs, acc, lane);
  f32x8 o[1];
  mlp_layer<1>(xs, w2, b2, 2, false, lane, o);
  int col = lane & 15, half = (lane >> 4) & 1;
  if (col < 3) {
#pragma unroll
    for (int v = 0; v < 8; ++v) {
      int row = half * 8 + v;
      delta[(size_t)(r0 + row) * 3 + col] = o[0][v];
    }
  }
}

// gnn edge: [f[s], c2[s]+delta[s]-c2[d]] (67) -> 64 relu -> 64 relu, seg-max over d
__global__ __launch_bounds__(BLOCK) void k_edge(const int* __restrict__ edges,
                                                const float* __restrict__ c2,
                                                const float* __restrict__ delta,
                                                const float* __restrict__ f,
                                                const unsigned* w1, const float* b1,
                                                const unsigned* w2, const float* b2,
                                                float* __restrict__ agg) {
  __shared__ float stage[WPB][16 * STG];
  __shared__ int sA[WPB][16], dA[WPB][16];
  int lane = threadIdx.x & 31, w = threadIdx.x >> 5;
  int e0 = (blockIdx.x * WPB + w) * 16;
  float* xs = stage[w];
  zero_stage<96>(xs, lane);
  if (lane < 16) {
    int e = e0 + lane;
    int s = edges[2 * e], d = edges[2 * e + 1];
    sA[w][lane] = s;
    dA[w][lane] = d;
#pragma unroll
    for (int i = 0; i < 3; ++i)
      xs[lane * STG + 64 + i] = c2[s * 3 + i] + delta[s * 3 + i] - c2[d * 3 + i];
  }
  for (int i = lane; i < 16 * 64; i += 32) {
    int row = i >> 6, k = i & 63;
    xs[row * STG + k] = f[(size_t)sA[w][row] * 64 + k];
  }
  f32x8 acc[4];
  mlp_layer<4>(xs, w1, b1, 3, true, lane, acc);
  frag_to_lds<4>(xs, acc, lane);
  mlp_layer<4>(xs, w2, b2, 2, true, lane, acc);
  int col = lane & 15, half = (lane >> 4) & 1;
#pragma unroll
  for (int j = 0; j < 4; ++j)
#pragma unroll
    for (int v = 0; v < 8; ++v) {
      int row = half * 8 + v;
      atomic_fmax_nonneg(&agg[(size_t)dA[w][row] * 64 + j * 16 + col], acc[j][v]);
    }
}

// gnn "g": MLP_g(agg) (64->64 relu ->64) + fprev => fout
__global__ __launch_bounds__(BLOCK) void k_g(const float* __restrict__ agg,
                                             const float* __restrict__ fprev,
                                             const unsigned* w1, const float* b1,
                                             const unsigned* w2, const float* b2,
                                             float* __restrict__ fout) {
  __shared__ float stage[WPB][16 * STG];
  int lane = threadIdx.x & 31, w = threadIdx.x >> 5;
  int r0 = (blockIdx.x * WPB + w) * 16;
  float* xs = stage[w];
  for (int i = lane; i < 16 * 64; i += 32) {
    int row = i >> 6, k = i & 63;
    xs[row * STG + k] = agg[(size_t)(r0 + row) * 64 + k];
  }
  f32x8 acc[4];
  mlp_layer<4>(xs, w1, b1, 2, true, lane, acc);
  frag_to_lds<4>(xs, acc, lane);
  mlp_layer<4>(xs, w2, b2, 2, false, lane, acc);
  int col = lane & 15, half = (lane >> 4) & 1;
#pragma unroll
  for (int j = 0; j < 4; ++j)
#pragma unroll
    for (int v = 0; v < 8; ++v) {
      size_t idx = (size_t)(r0 + half * 8 + v) * 64 + j * 16 + col;
      fout[idx] = acc[j][v] + fprev[idx];
    }
}

// l5: [f4[l2_labels], c1 - c2[l2_labels]] (67) -> 64 relu -> 64 relu => f5[N1,64]
__global__ __launch_bounds__(BLOCK) void k_l5(const float* __restrict__ f4,
                                              const float* __restrict__ c1,
                                              const float* __restrict__ c2,
                                              const int* __restrict__ lab2,
                                              const unsigned* w1, const float* b1,
                                              const unsigned* w2, const float* b2,
                                              float* __restrict__ f5) {
  __shared__ float stage[WPB][16 * STG];
  __shared__ int lab[WPB][16];
  int lane = threadIdx.x & 31, w = threadIdx.x >> 5;
  int r0 = (blockIdx.x * WPB + w) * 16;
  float* xs = stage[w];
  zero_stage<96>(xs, lane);
  if (lane < 16) {
    int r = r0 + lane;
    int lb = lab2[r];
    lab[w][lane] = lb;
#pragma unroll
    for (int i = 0; i < 3; ++i) xs[lane * STG + 64 + i] = c1[r * 3 + i] - c2[lb * 3 + i];
  }
  for (int i = lane; i < 16 * 64; i += 32) {
    int row = i >> 6, k = i & 63;
    xs[row * STG + k] = f4[(size_t)lab[w][row] * 64 + k];
  }
  f32x8 acc[4];
  mlp_layer<4>(xs, w1, b1, 3, true, lane, acc);
  frag_to_lds<4>(xs, acc, lane);
  mlp_layer<4>(xs, w2, b2, 2, true, lane, acc);
  int col = lane & 15, half = (lane >> 4) & 1;
#pragma unroll
  for (int j = 0; j < 4; ++j)
#pragma unroll
    for (int v = 0; v < 8; ++v)
      f5[(size_t)(r0 + half * 8 + v) * 64 + j * 16 + col] = acc[j][v];
}

// fused l7 + cls: [f5[l1_labels], points-c1[l1_labels]] -> (67->64->64 relu) -> (64->64 relu ->20)
__global__ __launch_bounds__(BLOCK) void k_l7cls(const float* __restrict__ f5,
                                                 const float* __restrict__ pts,
                                                 const float* __restrict__ c1,
                                                 const int* __restrict__ lab1,
                                                 const unsigned* w1a, const float* b1a,
                                                 const unsigned* w2a, const float* b2a,
                                                 const unsigned* w1c, const float* b1c,
                                                 const unsigned* w2c, const float* b2c,
                                                 float* __restrict__ out) {
  __shared__ float stage[WPB][16 * STG];
  __shared__ int lab[WPB][16];
  int lane = threadIdx.x & 31, w = threadIdx.x >> 5;
  int r0 = (blockIdx.x * WPB + w) * 16;
  float* xs = stage[w];
  zero_stage<96>(xs, lane);
  if (lane < 16) {
    int r = r0 + lane;
    int lb = lab1[r];
    lab[w][lane] = lb;
#pragma unroll
    for (int i = 0; i < 3; ++i) xs[lane * STG + 64 + i] = pts[r * 3 + i] - c1[lb * 3 + i];
  }
  for (int i = lane; i < 16 * 64; i += 32) {
    int row = i >> 6, k = i & 63;
    xs[row * STG + k] = f5[(size_t)lab[w][row] * 64 + k];
  }
  f32x8 acc[4];
  mlp_layer<4>(xs, w1a, b1a, 3, true, lane, acc);  // l7 layer 1
  frag_to_lds<4>(xs, acc, lane);
  mlp_layer<4>(xs, w2a, b2a, 2, true, lane, acc);  // l7 layer 2
  frag_to_lds<4>(xs, acc, lane);
  mlp_layer<4>(xs, w1c, b1c, 2, true, lane, acc);  // cls hidden
  frag_to_lds<4>(xs, acc, lane);
  f32x8 o[2];
  mlp_layer<2>(xs, w2c, b2c, 2, false, lane, o);   // cls out (20, padded to 32)
  int col = lane & 15, half = (lane >> 4) & 1;
#pragma unroll
  for (int j = 0; j < 2; ++j) {
    int n = j * 16 + col;
    if (n < NCLS) {
#pragma unroll
      for (int v = 0; v < 8; ++v)
        out[(size_t)(r0 + half * 8 + v) * NCLS + n] = o[j][v];
    }
  }
}

// =====================================================================
extern "C" void kernel_launch(void* const* d_in, const int* in_sizes, int n_in,
                              void* d_out, int out_size, void* d_ws, size_t ws_size,
                              hipStream_t stream) {
  (void)in_sizes; (void)n_in; (void)out_size; (void)ws_size;
  const float* rem  = (const float*)d_in[0];
  const float* pts  = (const float*)d_in[1];
  const float* c1   = (const float*)d_in[2];
  const float* c2   = (const float*)d_in[3];
  const int* l2e    = (const int*)d_in[5];
  const int* lab1   = (const int*)d_in[6];
  const int* lab2   = (const int*)d_in[7];
  // params flattened in _SPEC order, 4 per name: w1, w2, b1, b2
  auto P = [&](int name, int which) -> const float* {
    return (const float*)d_in[8 + name * 4 + which];
  };
  // name indices: l1=0 l3=1 g4_h=2 g4_f=3 g4_g=4 g41_h=5 g41_f=6 g41_g=7 l5=8 l7=9 cls=10
  static const int DIN[11]  = {4, 67, 64, 67, 64, 64, 67, 64, 67, 67, 64};
  static const int DOUT[11] = {64, 64, 3, 64, 64, 3, 64, 64, 64, 64, 20};

  float* ws = (float*)d_ws;
  size_t off = 0;
  float* f1    = ws + off; off += (size_t)N1 * 64;
  float* f2    = ws + off; off += (size_t)N2 * 64;
  float* delta = ws + off; off += (size_t)N2 * 4;
  float* agg4  = ws + off; off += (size_t)N2 * 64;
  float* f4    = ws + off; off += (size_t)N2 * 64;
  float* agg41 = ws + off; off += (size_t)N2 * 64;
  float* f4b   = ws + off; off += (size_t)N2 * 64;
  float* f5    = ws + off; off += (size_t)N1 * 64;
  float* out   = (float*)d_out;

  // ---- pack region: bf16 weight fragments + padded biases ----
  unsigned* W1f[11]; float* B1p[11]; unsigned* W2f[11]; float* B2p[11];
  int KT1[11], NT2[11];
  {
    unsigned* cur = (unsigned*)(ws + off);
    for (int i = 0; i < 11; ++i) {
      KT1[i] = (DIN[i] + 31) / 32;
      NT2[i] = (DOUT[i] + 15) / 16;
      W1f[i] = cur; cur += (size_t)KT1[i] * 4 * 256;       // layer1: NT=4 (dh=64)
      W2f[i] = cur; cur += (size_t)2 * NT2[i] * 256;       // layer2: KT=2 (dh=64)
      B1p[i] = (float*)cur; cur += 64;                     // padded to 64
      B2p[i] = (float*)cur; cur += (size_t)NT2[i] * 16;    // padded to NT2*16
    }
  }

  auto fill0 = [&](float* p, int n) {
    k_fill<<<(n + 255) / 256, 256, 0, stream>>>(p, n, 0.f);
  };
  fill0(f1, N1 * 64);
  fill0(f2, N2 * 64);
  fill0(agg4, N2 * 64);
  fill0(agg41, N2 * 64);

  for (int i = 0; i < 11; ++i) {
    int t1 = KT1[i] * 4 * 256;
    k_packB<<<(t1 + 255) / 256, 256, 0, stream>>>(P(i, 0), DIN[i], 64, KT1[i], 4, W1f[i]);
    int t2 = 2 * NT2[i] * 256;
    k_packB<<<(t2 + 255) / 256, 256, 0, stream>>>(P(i, 1), 64, DOUT[i], 2, NT2[i], W2f[i]);
    k_packbias<<<1, 64, 0, stream>>>(P(i, 2), 64, B1p[i]);
    k_packbias<<<1, NT2[i] * 16, 0, stream>>>(P(i, 3), DOUT[i], B2p[i]);
  }

  k_l1<<<N / 16 / WPB, BLOCK, 0, stream>>>(rem, pts, c1, lab1,
                                           W1f[0], B1p[0], W2f[0], B2p[0], f1);
  k_l3<<<N1 / 16 / WPB, BLOCK, 0, stream>>>(f1, c1, c2, lab2,
                                            W1f[1], B1p[1], W2f[1], B2p[1], f2);
  // gnn pass 1 (g4)
  k_delta<<<N2 / 16 / WPB, BLOCK, 0, stream>>>(f2, W1f[2], B1p[2], W2f[2], B2p[2], delta);
  k_edge<<<E2 / 16 / WPB, BLOCK, 0, stream>>>(l2e, c2, delta, f2,
                                              W1f[3], B1p[3], W2f[3], B2p[3], agg4);
  k_g<<<N2 / 16 / WPB, BLOCK, 0, stream>>>(agg4, f2, W1f[4], B1p[4], W2f[4], B2p[4], f4);
  // gnn pass 2 (g41)
  k_delta<<<N2 / 16 / WPB, BLOCK, 0, stream>>>(f4, W1f[5], B1p[5], W2f[5], B2p[5], delta);
  k_edge<<<E2 / 16 / WPB, BLOCK, 0, stream>>>(l2e, c2, delta, f4,
                                              W1f[6], B1p[6], W2f[6], B2p[6], agg41);
  k_g<<<N2 / 16 / WPB, BLOCK, 0, stream>>>(agg41, f4, W1f[7], B1p[7], W2f[7], B2p[7], f4b);

  k_l5<<<N1 / 16 / WPB, BLOCK, 0, stream>>>(f4b, c1, c2, lab2,
                                            W1f[8], B1p[8], W2f[8], B2p[8], f5);
  k_l7cls<<<N / 16 / WPB, BLOCK, 0, stream>>>(f5, pts, c1, lab1,
                                              W1f[9], B1p[9], W2f[9], B2p[9],
                                              W1f[10], B1p[10], W2f[10], B2p[10], out);
}
